// GNN_28028956574153
// MI455X (gfx1250) — matrix operand order repaired and checked by reading, not compiled, not run on hardware
//
#include <hip/hip_runtime.h>

// ---------------------------------------------------------------------------
// Problem constants (match reference)
// ---------------------------------------------------------------------------
#define Bc 32
#define Nn 512
#define Cc 256
#define Kk 16
#define NNODE (Bc * Nn)          // 16384
#define NEDGE (NNODE * Kk)       // 262144
#define L0_IN (2 * Cc + 7)       // 519
#define KPAD0 544                // 519 padded up to multiple of 32
#define KPAD12 512
#define MS 552                   // LDS row stride in f16 (16B aligned, conflict-free)
#define EPSV 1e-5f

typedef _Float16 v16h __attribute__((ext_vector_type(16)));
typedef _Float16 h8   __attribute__((ext_vector_type(8)));
typedef float    v8f  __attribute__((ext_vector_type(8)));

union FragH { v16h v; uint4 q[2]; };

// ---------------------------------------------------------------------------
// Utility kernels
// ---------------------------------------------------------------------------
__global__ void gnn_zero_f32(float* __restrict__ p, int n) {
  int i = blockIdx.x * blockDim.x + threadIdx.x;
  if (i < n) p[i] = 0.0f;
}

// W [Kin][256] f32 row-major  ->  wT [256][KW] f16 (zero padded rows >= Kin)
__global__ void gnn_transpose_cvt(const float* __restrict__ W,
                                  _Float16* __restrict__ wT, int Kin, int KW) {
  int i = blockIdx.x * blockDim.x + threadIdx.x;
  if (i >= 256 * KW) return;
  int n = i / KW, k = i - n * KW;
  wT[i] = (k < Kin) ? (_Float16)W[k * 256 + n] : (_Float16)0.0f;
}

// ---------------------------------------------------------------------------
// Global-info MLP:  h0[n][c] = cat(pf[n], box[n]) . Wg[:,c] + bg[c]
// ---------------------------------------------------------------------------
__global__ __launch_bounds__(256)
void gnn_mlp0(const float* __restrict__ pf, const float* __restrict__ rois,
              const float* __restrict__ Wg, const float* __restrict__ bg,
              float* __restrict__ h0) {
  __shared__ float sin_[Cc + 7];
  int n = blockIdx.x, t = threadIdx.x;
  sin_[t] = pf[n * Cc + t];
  if (t < 7) sin_[Cc + t] = rois[n * 7 + t];
  __syncthreads();
  float acc = bg[t];
  for (int d = 0; d < Cc + 7; ++d)
    acc = fmaf(sin_[d], Wg[d * 256 + t], acc);
  h0[n * Cc + t] = acc;
}

// per-channel mean / rstd over R rows of h [R][256]; one block per channel
__global__ __launch_bounds__(256)
void gnn_colstats(const float* __restrict__ h, int R,
                  float* __restrict__ mu, float* __restrict__ rstd) {
  __shared__ float s1[256], s2[256];
  int c = blockIdx.x, t = threadIdx.x;
  float a = 0.f, b = 0.f;
  for (int r = t; r < R; r += 256) {
    float v = h[(size_t)r * 256 + c];
    a += v; b += v * v;
  }
  s1[t] = a; s2[t] = b;
  __syncthreads();
  for (int s = 128; s > 0; s >>= 1) {
    if (t < s) { s1[t] += s1[t + s]; s2[t] += s2[t + s]; }
    __syncthreads();
  }
  if (t == 0) {
    float m = s1[0] / (float)R;
    float var = s2[0] / (float)R - m * m;
    mu[c] = m;
    rstd[c] = rsqrtf(var + EPSV);
  }
}

// BN + ReLU for x0; writes f32 into out cols [0,256) and an f16 copy
__global__ void gnn_bn_relu_x0(const float* __restrict__ h0,
                               const float* __restrict__ mu,
                               const float* __restrict__ rstd,
                               const float* __restrict__ g,
                               const float* __restrict__ beta,
                               float* __restrict__ out,
                               _Float16* __restrict__ x16) {
  int i = blockIdx.x * blockDim.x + threadIdx.x;
  if (i >= NNODE * Cc) return;
  int n = i >> 8, c = i & 255;
  float v = (h0[i] - mu[c]) * rstd[c] * g[c] + beta[c];
  v = fmaxf(v, 0.f);
  out[(size_t)n * 1024 + c] = v;
  x16[i] = (_Float16)v;
}

// convert one 256-col stripe of out back to f16 node features
__global__ void gnn_cvt_x16(const float* __restrict__ out, int colOff,
                            _Float16* __restrict__ x16) {
  int i = blockIdx.x * blockDim.x + threadIdx.x;
  if (i >= NNODE * Cc) return;
  int n = i >> 8, c = i & 255;
  x16[i] = (_Float16)out[(size_t)n * 1024 + colOff + c];
}

// mu/rstd from accumulated column sums over E edges
__global__ void gnn_finalize_stats(const float* __restrict__ cs,
                                   const float* __restrict__ csq,
                                   float* __restrict__ mu,
                                   float* __restrict__ rstd) {
  int c = threadIdx.x;
  float m = cs[c] / (float)NEDGE;
  float var = csq[c] / (float)NEDGE - m * m;
  mu[c] = m;
  rstd[c] = rsqrtf(var + EPSV);
}

// ---------------------------------------------------------------------------
// EdgeConv GEMM (WMMA f32_16x16x32_f16), software-pipelined
//   block: 512 threads (16 wave32), 64 edges x 256 outputs per block
//   mode 0: accumulate per-channel sum / sumsq of h = m@W + b
//   mode 1: recompute h, BN+ReLU, scatter-max into out[:, colOff..colOff+256)
// ---------------------------------------------------------------------------
__global__ __launch_bounds__(512)
void gnn_edge_gemm(const _Float16* __restrict__ x16,
                   const float* __restrict__ rois,
                   const int* __restrict__ esrc, const int* __restrict__ edst,
                   const _Float16* __restrict__ wT,
                   const float* __restrict__ bias,
                   int mode, int KW, int useBox,
                   float* __restrict__ colsum, float* __restrict__ colsumsq,
                   const float* __restrict__ mu, const float* __restrict__ rstd,
                   const float* __restrict__ gamma, const float* __restrict__ beta,
                   float* __restrict__ out, int outColOff) {
  __shared__ _Float16 smsg[64 * MS];
  __shared__ int ssrc[64], sdst[64];

  const int tid = threadIdx.x;
  const int e0 = blockIdx.x * 64;

  if (tid < 64) {
    ssrc[tid] = esrc[e0 + tid];
    sdst[tid] = edst[e0 + tid];
  }
  // hint the weight panel into cache (gfx1250 global_prefetch_b8)
  if (tid < 256) __builtin_prefetch(wT + (size_t)tid * KW, 0, 1);
  __syncthreads();

  // ---- build the padded edge-message tile in LDS (f16) ----
  // part 1: d in [0,256)  : x_j - x_i, 8-wide f16 vectors
  for (int idx = tid; idx < 64 * 32; idx += 512) {
    int e = idx >> 5, g = idx & 31, d = g << 3;
    int s = ssrc[e], dn = sdst[e];
    h8 xs = *reinterpret_cast<const h8*>(x16 + (size_t)s * 256 + d);
    h8 xd = *reinterpret_cast<const h8*>(x16 + (size_t)dn * 256 + d);
    *reinterpret_cast<h8*>(&smsg[e * MS + d]) = xs - xd;
  }
  // part 2: d in [256,512): x_i row copy, async global -> LDS (ASYNCcnt path)
  for (int idx = tid; idx < 64 * 32; idx += 512) {
    int e = idx >> 5, g = idx & 31;
    int dn = sdst[e];
    unsigned lds_off = (unsigned)(size_t)&smsg[e * MS + 256 + (g << 3)];
    const _Float16* gp = x16 + (size_t)dn * 256 + (g << 3);
    asm volatile("global_load_async_to_lds_b128 %0, %1, off"
                 :: "v"(lds_off), "v"(gp) : "memory");
  }
  // part 3: layer-0 tail d in [512,544): box deltas + zero pad
  if (KW > 512) {
    for (int idx = tid; idx < 64 * 32; idx += 512) {
      int e = idx >> 5, d = 512 + (idx & 31);
      int s = ssrc[e], dn = sdst[e];
      _Float16 v = (_Float16)0.0f;
      if (useBox && d < 519) {
        int b = d - 512;
        v = (_Float16)(rois[s * 7 + b] - rois[dn * 7 + b]);
      }
      smsg[e * MS + d] = v;
    }
  }
  asm volatile("s_wait_asynccnt 0x0" ::: "memory");
  __syncthreads();

  const int wid = tid >> 5, lane = tid & 31;
  const int m = lane & 15;
  const bool lowHalf = lane < 16;
  const int rt = wid & 3;        // row tile 0..3 (16 edges each)
  const int qg = wid >> 2;       // col group 0..3 (4 col tiles each)

  const int rowBase = (rt * 16 + m) * MS;
  const _Float16* wTb[4];
#pragma unroll
  for (int t = 0; t < 4; ++t)
    wTb[t] = wT + (size_t)((qg * 4 + t) * 16 + m) * KW;

  v8f acc[4] = {};
  FragH A0, A1, B0[4], B1[4];

  auto loadA = [&](FragH& A, int ks) {
    const int h0s = ks + (lowHalf ? 0 : 16);
    const int h1s = ks + (lowHalf ? 8 : 24);
    A.q[0] = *reinterpret_cast<const uint4*>(&smsg[rowBase + h0s]);
    A.q[1] = *reinterpret_cast<const uint4*>(&smsg[rowBase + h1s]);
  };
  auto loadB = [&](FragH* Bf, int ks) {
    const int kb = ks + (lowHalf ? 0 : 16);
#pragma unroll
    for (int t = 0; t < 4; ++t) {
      const uint4* bp = reinterpret_cast<const uint4*>(wTb[t] + kb);
      Bf[t].q[0] = bp[0];
      Bf[t].q[1] = bp[1];
    }
  };
  auto mm = [&](FragH& A, FragH* Bf) {
#pragma unroll
    for (int t = 0; t < 4; ++t)
      acc[t] = __builtin_amdgcn_wmma_f32_16x16x32_f16(
          false, A.v, false, Bf[t].v, (short)0, acc[t], false, false);
  };

  // double-buffered software pipeline over K (KW/32 = 16 or 17 steps)
  loadA(A0, 0);
  loadB(B0, 0);
  int ks = 0;
  for (;;) {
    int ksn = ks + 32;
    bool more = ksn < KW;
    if (more) { loadA(A1, ksn); loadB(B1, ksn); }
    mm(A0, B0);
    ks = ksn;
    if (!more) break;
    ksn = ks + 32;
    more = ksn < KW;
    if (more) { loadA(A0, ksn); loadB(B0, ksn); }
    mm(A1, B1);
    ks = ksn;
    if (!more) break;
  }

  if (mode == 0) {
#pragma unroll
    for (int t = 0; t < 4; ++t) {
      const int n = (qg * 4 + t) * 16 + m;
      const float bz = bias[n];
      float s = 0.f, s2 = 0.f;
#pragma unroll
      for (int r = 0; r < 8; ++r) {
        float h = acc[t][r] + bz;
        s += h; s2 += h * h;
      }
      s  += __shfl_xor(s, 16, 32);
      s2 += __shfl_xor(s2, 16, 32);
      if (lowHalf) {
        atomicAdd(&colsum[n], s);
        atomicAdd(&colsumsq[n], s2);
      }
    }
  } else {
#pragma unroll
    for (int t = 0; t < 4; ++t) {
      const int n = (qg * 4 + t) * 16 + m;
      const float bz = bias[n], muv = mu[n], rs = rstd[n];
      const float g = gamma[n], bt = beta[n];
#pragma unroll
      for (int r = 0; r < 8; ++r) {
        float h = acc[t][r] + bz;
        float hv = fmaxf((h - muv) * rs * g + bt, 0.f);
        const int M = lowHalf ? r : (r + 8);
        const int dn = sdst[rt * 16 + M];
        // post-ReLU values are >= 0: uint ordering == float ordering
        atomicMax(reinterpret_cast<unsigned int*>(
                      &out[(size_t)dn * 1024 + outColOff + n]),
                  __float_as_uint(hv));
      }
    }
  }
}

// ---------------------------------------------------------------------------
// Host launcher
// ---------------------------------------------------------------------------
extern "C" void kernel_launch(void* const* d_in, const int* in_sizes, int n_in,
                              void* d_out, int out_size, void* d_ws, size_t ws_size,
                              hipStream_t stream) {
  (void)in_sizes; (void)n_in; (void)out_size; (void)ws_size;

  const float* pf    = (const float*)d_in[0];
  const float* rois  = (const float*)d_in[1];
  const int*   esrc  = (const int*)  d_in[2];
  const int*   edst  = (const int*)  d_in[3];
  const float* Wg    = (const float*)d_in[4];
  const float* bg    = (const float*)d_in[5];
  const float* gg    = (const float*)d_in[6];
  const float* betag = (const float*)d_in[7];
  const float* Wl[3]    = { (const float*)d_in[8],  (const float*)d_in[12], (const float*)d_in[16] };
  const float* bl[3]    = { (const float*)d_in[9],  (const float*)d_in[13], (const float*)d_in[17] };
  const float* gl[3]    = { (const float*)d_in[10], (const float*)d_in[14], (const float*)d_in[18] };
  const float* betal[3] = { (const float*)d_in[11], (const float*)d_in[15], (const float*)d_in[19] };

  float* out = (float*)d_out;

  // workspace carve-up (256B aligned chunks, ~26 MB total)
  char* w = (char*)d_ws;
  size_t off = 0;
  auto take = [&](size_t bytes) {
    char* p = w + off;
    off += (bytes + 255) & ~(size_t)255;
    return p;
  };
  _Float16* w0T = (_Float16*)take((size_t)256 * KPAD0  * 2);
  _Float16* w1T = (_Float16*)take((size_t)256 * KPAD12 * 2);
  _Float16* w2T = (_Float16*)take((size_t)256 * KPAD12 * 2);
  _Float16* x16 = (_Float16*)take((size_t)NNODE * Cc * 2);
  float*    h0  = (float*)   take((size_t)NNODE * Cc * 4);
  float*    cs  = (float*)   take(256 * 4);
  float*    csq = (float*)   take(256 * 4);
  float*    mu  = (float*)   take(256 * 4);
  float*    rs  = (float*)   take(256 * 4);
  _Float16* wT[3] = { w0T, w1T, w2T };
  const int KW[3] = { KPAD0, KPAD12, KPAD12 };

  // zero entire output (scatter-max baseline + x0 region overwritten later)
  {
    int n = NNODE * 1024;
    gnn_zero_f32<<<(n + 255) / 256, 256, 0, stream>>>(out, n);
  }

  // pre-transpose + f16-convert the three EdgeConv weights
  for (int L = 0; L < 3; ++L) {
    int n = 256 * KW[L];
    gnn_transpose_cvt<<<(n + 255) / 256, 256, 0, stream>>>(
        Wl[L], wT[L], (L == 0) ? L0_IN : 2 * Cc, KW[L]);
  }

  // global-info MLP -> x0
  gnn_mlp0<<<NNODE, 256, 0, stream>>>(pf, rois, Wg, bg, h0);
  gnn_colstats<<<256, 256, 0, stream>>>(h0, NNODE, mu, rs);
  gnn_bn_relu_x0<<<(NNODE * Cc) / 256, 256, 0, stream>>>(
      h0, mu, rs, gg, betag, out, x16);

  // three EdgeConv layers (stats pass + finalize pass, recompute GEMM)
  for (int L = 0; L < 3; ++L) {
    gnn_zero_f32<<<2, 256, 0, stream>>>(cs, 512);  // cs + csq contiguous
    gnn_edge_gemm<<<NEDGE / 64, 512, 0, stream>>>(
        x16, rois, esrc, edst, wT[L], bl[L],
        /*mode=*/0, KW[L], /*useBox=*/(L == 0) ? 1 : 0,
        cs, csq, mu, rs, gl[L], betal[L], out, (L + 1) * 256);
    gnn_finalize_stats<<<1, 256, 0, stream>>>(cs, csq, mu, rs);
    gnn_edge_gemm<<<NEDGE / 64, 512, 0, stream>>>(
        x16, rois, esrc, edst, wT[L], bl[L],
        /*mode=*/1, KW[L], /*useBox=*/(L == 0) ? 1 : 0,
        cs, csq, mu, rs, gl[L], betal[L], out, (L + 1) * 256);
    if (L < 2) {
      gnn_cvt_x16<<<(NNODE * Cc) / 256, 256, 0, stream>>>(
          out, (L + 1) * 256, x16);
    }
  }
}
